// MOLERouterV3_49529562858338
// MI455X (gfx1250) — matrix operand order repaired and checked by reading, not compiled, not run on hardware
//
#include <hip/hip_runtime.h>
#include <hip/hip_bf16.h>

typedef float  v2f  __attribute__((ext_vector_type(2)));
typedef float  v4f  __attribute__((ext_vector_type(4)));
typedef float  v8f  __attribute__((ext_vector_type(8)));
typedef __bf16 v2bf __attribute__((ext_vector_type(2)));
typedef __bf16 v8bf __attribute__((ext_vector_type(8)));
typedef __bf16 v16bf __attribute__((ext_vector_type(16)));

#define N_TOK   131072
#define IN_F    512
#define HID     128
#define NEXP    48
#define TOPK    6
#define ROWS    64      // rows per block (4 waves x 16)
#define KC      64      // K chunk of W1 staged in LDS
#define S1B     72      // LDS stride (bf16 elems) for split-W1T rows (KC + 8 pad)
#define SH      132     // LDS stride for h rows (HID + 4 pad)
#define SS      52      // LDS stride for score rows (NEXP + 4 pad)

__device__ __forceinline__ float fast_sigmoid(float x) {
    return 1.0f / (1.0f + __expf(-x));
}

union __align__(16) SmemU {
    struct {                                    // GEMM1 phase: split W1^T chunk
        __bf16 hi[HID * S1B];                   // 18,432 B
        __bf16 lo[HID * S1B];                   // 18,432 B
    } w;
    struct {                                    // after GEMM1
        float h[ROWS * SH];                     // 33,792 B
        float s[ROWS * SS];                     // 13,312 B
    } p;
};

__global__ void router_init(unsigned* __restrict__ g_load) {
    int i = threadIdx.x;
    if (i < NEXP) g_load[i] = 0u;
}

__global__ __launch_bounds__(128)
void router_main(const float* __restrict__ X,  const float* __restrict__ W1,
                 const float* __restrict__ B1, const float* __restrict__ W2,
                 const float* __restrict__ B2, const float* __restrict__ EBIAS,
                 float* __restrict__ coeffs, unsigned* __restrict__ g_load,
                 float* __restrict__ g_mon)
{
    __shared__ SmemU sm;
    __shared__ unsigned cnt[NEXP];
    __shared__ float msum[ROWS];

    const int tid  = threadIdx.x;
    const int lane = tid & 31;
    const int wv   = tid >> 5;              // wave id 0..3
    const int m16  = lane & 15;
    const int hi   = lane >> 4;             // 0 or 1 (lane half)
    const int kb   = hi * 2;                // K sub-offset for the f32 x4 frags
    const int row0 = blockIdx.x * ROWS;

    if (tid < NEXP) cnt[tid] = 0u;

    // ------- GEMM1: h = silu(X @ W1 + b1), split-bf16 WMMA 16x16x32 ---------
    v8f acc[8];
    #pragma unroll
    for (int j = 0; j < 8; ++j)
        acc[j] = v8f{0.f,0.f,0.f,0.f,0.f,0.f,0.f,0.f};

    const float* xrow = X + (size_t)(row0 + wv * 16 + m16) * IN_F;

    for (int k0 = 0; k0 < IN_F; k0 += KC) {
        __syncthreads();   // previous chunk fully consumed before overwrite
        // Cooperative transpose + hi/lo split of W1[k0:k0+KC, :] into LDS
        {
            const int n = tid;   // 0..127 == HID
            #pragma unroll 2
            for (int k = 0; k < KC; k += 2) {
                float x0 = W1[(size_t)(k0 + k    ) * HID + n];
                float x1 = W1[(size_t)(k0 + k + 1) * HID + n];
                __bf16 h0 = (__bf16)x0, h1 = (__bf16)x1;
                __bf16 l0 = (__bf16)(x0 - (float)h0);
                __bf16 l1 = (__bf16)(x1 - (float)h1);
                *(v2bf*)&sm.w.hi[n * S1B + k] = v2bf{h0, h1};
                *(v2bf*)&sm.w.lo[n * S1B + k] = v2bf{l0, l1};
            }
        }
        __syncthreads();

        if (k0 + KC < IN_F)                       // global_prefetch_b8 flavor
            __builtin_prefetch(xrow + k0 + KC, 0, 0);

        #pragma unroll
        for (int kc = 0; kc < KC; kc += 32) {
            // A fragment: this lane owns row m16 and
            // K = kc + {16h..16h+15} (contiguous); load fp32, split in-reg.
            const float* xp = xrow + k0 + kc;
            v4f xa[4];
            xa[0] = *(const v4f*)(xp + 16 * hi);
            xa[1] = *(const v4f*)(xp + 16 * hi + 4);
            xa[2] = *(const v4f*)(xp + 16 * hi + 8);
            xa[3] = *(const v4f*)(xp + 16 * hi + 12);
            v16bf ah, al;
            #pragma unroll
            for (int q = 0; q < 4; ++q) {
                #pragma unroll
                for (int t = 0; t < 4; ++t) {
                    float x = xa[q][t];
                    __bf16 hh = (__bf16)x;
                    ah[4 * q + t] = hh;
                    al[4 * q + t] = (__bf16)(x - (float)hh);
                }
            }
            #pragma unroll
            for (int j = 0; j < 8; ++j) {
                const int n   = m16 + 16 * j;
                const int off = n * S1B + kc + 16 * hi;   // 16 contiguous K
                v8bf bh0 = *(const v8bf*)&sm.w.hi[off];
                v8bf bh1 = *(const v8bf*)&sm.w.hi[off + 8];
                v8bf bl0 = *(const v8bf*)&sm.w.lo[off];
                v8bf bl1 = *(const v8bf*)&sm.w.lo[off + 8];
                v16bf bh = __builtin_shufflevector(bh0, bh1,
                            0,1,2,3,4,5,6,7,8,9,10,11,12,13,14,15);
                v16bf bl = __builtin_shufflevector(bl0, bl1,
                            0,1,2,3,4,5,6,7,8,9,10,11,12,13,14,15);
                // acc += Ahi*Bhi + Ahi*Blo + Alo*Bhi   (fp32-class precision)
                acc[j] = __builtin_amdgcn_wmma_f32_16x16x32_bf16(
                            false, ah, false, bh, (short)0, acc[j], false, false);
                acc[j] = __builtin_amdgcn_wmma_f32_16x16x32_bf16(
                            false, ah, false, bl, (short)0, acc[j], false, false);
                acc[j] = __builtin_amdgcn_wmma_f32_16x16x32_bf16(
                            false, al, false, bh, (short)0, acc[j], false, false);
            }
        }
    }

    __syncthreads();   // all waves done reading w1 split; region re-used for h

    // Epilogue: bias + SiLU, stage h in LDS (row-major, A-fragment friendly)
    #pragma unroll
    for (int j = 0; j < 8; ++j) {
        const int n  = m16 + 16 * j;
        const float bj = B1[n];
        #pragma unroll
        for (int r = 0; r < 8; ++r) {
            const int mrow = wv * 16 + r + 8 * hi;
            float v = acc[j][r] + bj;
            sm.p.h[mrow * SH + n] = v * fast_sigmoid(v);
        }
    }
    __syncthreads();

    // ------- GEMM2: logits = h @ W2 + b2, fp32 WMMA 16x16x4 -----------------
    v8f acc2[3];
    #pragma unroll
    for (int j = 0; j < 3; ++j)
        acc2[j] = v8f{0.f,0.f,0.f,0.f,0.f,0.f,0.f,0.f};

    for (int ks = 0; ks < HID; ks += 4) {
        v2f a = *(const v2f*)&sm.p.h[(wv * 16 + m16) * SH + ks + kb];
        #pragma unroll
        for (int j = 0; j < 3; ++j) {
            const int n = m16 + 16 * j;           // expert 0..47
            v2f b = { W2[(size_t)(ks + kb    ) * NEXP + n],
                      W2[(size_t)(ks + kb + 1) * NEXP + n] };
            acc2[j] = __builtin_amdgcn_wmma_f32_16x16x4_f32(
                        false, a, false, b, (short)0, acc2[j], false, false);
        }
    }

    // scores = sigmoid(logits + b2) -> LDS
    #pragma unroll
    for (int j = 0; j < 3; ++j) {
        const int n  = m16 + 16 * j;
        const float bj = B2[n];
        #pragma unroll
        for (int r = 0; r < 8; ++r) {
            const int mrow = wv * 16 + r + 8 * hi;
            sm.p.s[mrow * SS + n] = fast_sigmoid(acc2[j][r] + bj);
        }
    }
    __syncthreads();

    // ------- per-token biased top-6, normalize ------------------------------
    if (tid < ROWS) {
        float sc[NEXP], bi[NEXP];
        for (int e = 0; e < NEXP; ++e) {
            sc[e] = sm.p.s[tid * SS + e];
            bi[e] = sc[e] + EBIAS[e];
        }
        int sel[TOPK];
        float ssum = 0.0f, smax = 0.0f;
        for (int k = 0; k < TOPK; ++k) {
            float best = -1e30f; int be = 0;
            for (int e = 0; e < NEXP; ++e)
                if (bi[e] > best) { best = bi[e]; be = e; }   // ties -> lowest idx
            bi[be] = -1e30f;
            sel[k] = be;
            ssum += sc[be];
            smax = fmaxf(smax, sc[be]);
        }
        const float inv = 1.0f / (ssum + 1e-8f);
        msum[tid] = smax * inv;

        float pr[TOPK];
        for (int k = 0; k < TOPK; ++k) pr[k] = sc[sel[k]] * inv;
        for (int e = 0; e < NEXP; ++e) sm.p.s[tid * SS + e] = 0.0f;
        for (int k = 0; k < TOPK; ++k) {
            sm.p.s[tid * SS + sel[k]] = pr[k];
            atomicAdd(&cnt[sel[k]], 1u);
        }
    }
    __syncthreads();

    // coalesced dense coeff write: 64 rows x 12 float4
    for (int i = tid; i < ROWS * 12; i += blockDim.x) {
        const int rr = i / 12, c4 = i % 12;
        v4f v = *(v4f*)&sm.p.s[rr * SS + c4 * 4];
        *(v4f*)&coeffs[(size_t)(row0 + rr) * NEXP + c4 * 4] = v;
    }
    if (tid == 0) {                        // fixed-order partial (deterministic)
        float s = 0.0f;
        for (int i = 0; i < ROWS; ++i) s += msum[i];
        g_mon[blockIdx.x] = s;
    }
    if (tid < NEXP && cnt[tid]) atomicAdd(&g_load[tid], cnt[tid]);  // integer: exact
}

__global__ __launch_bounds__(128)
void router_final(const unsigned* __restrict__ g_load,
                  const float* __restrict__ g_mon, int nblk,
                  const float* __restrict__ ema,
                  float* __restrict__ out_mon, float* __restrict__ out_cv)
{
    __shared__ float ps[128];
    const int tid = threadIdx.x;
    float s = 0.0f;
    for (int i = tid; i < nblk; i += 128) s += g_mon[i];
    ps[tid] = s;
    __syncthreads();
    if (tid == 0) {
        float tot = 0.0f;
        for (int i = 0; i < 128; ++i) tot += ps[i];
        *out_mon = tot / (float)N_TOK;

        float em[NEXP], mean = 0.0f;
        for (int e = 0; e < NEXP; ++e) {
            em[e] = ema[e] * 0.9f + (float)g_load[e] * 0.1f;
            mean += em[e];
        }
        mean /= (float)NEXP;
        float var = 0.0f;
        for (int e = 0; e < NEXP; ++e) {
            float d = em[e] - mean;
            var += d * d;
        }
        var /= (float)(NEXP - 1);                 // Bessel (ddof=1), as torch .std()
        *out_cv = sqrtf(var) / (mean + 1e-8f);
    }
}

extern "C" void kernel_launch(void* const* d_in, const int* in_sizes, int n_in,
                              void* d_out, int out_size, void* d_ws, size_t ws_size,
                              hipStream_t stream) {
    const float* X   = (const float*)d_in[0];
    const float* W1  = (const float*)d_in[1];
    const float* b1  = (const float*)d_in[2];
    const float* W2  = (const float*)d_in[3];
    const float* b2  = (const float*)d_in[4];
    const float* eb  = (const float*)d_in[5];
    const float* ema = (const float*)d_in[6];

    float* coeffs   = (float*)d_out;
    float* out_mon  = coeffs + (size_t)N_TOK * NEXP;
    float* out_cv   = out_mon + 1;

    unsigned* g_load = (unsigned*)d_ws;                    // 48 u32
    float*    g_mon  = (float*)((char*)d_ws + 256);        // 2048 f32 partials

    const int nblk = N_TOK / ROWS;                         // 2048

    router_init<<<1, 64, 0, stream>>>(g_load);
    router_main<<<nblk, 128, 0, stream>>>(X, W1, b1, W2, b2, eb,
                                          coeffs, g_load, g_mon);
    router_final<<<1, 128, 0, stream>>>(g_load, g_mon, nblk, ema,
                                        out_mon, out_cv);
}
